// classifier_network5_24086176596023
// MI455X (gfx1250) — matrix operand order repaired and checked by reading, not compiled, not run on hardware
//
#include <hip/hip_runtime.h>
#include <math.h>

typedef __attribute__((ext_vector_type(16))) _Float16 v16h;
typedef __attribute__((ext_vector_type(8)))  _Float16 v8h;
typedef __attribute__((ext_vector_type(8)))  float    v8f;

#define N_ROWS 8192
#define DDIM   768
#define LDS_STRIDE 776            // 768 + 8 halves pad -> conflict-free ds_load_b128
#define TEMP_SCALE 28.853900817779268f   // (1/0.05) * log2(e)
#define LN2F 0.6931471805599453f
#define MAXLOG 30.0f              // fixed LSE shift: |logit|<=28.86 since |cos|<=1

// ---------------------------------------------------------------------------
// Kernel 1: row-normalize f32 inputs -> f16 matrices in workspace.
// ---------------------------------------------------------------------------
__global__ __launch_bounds__(256)
void normalize_kernel(const float* __restrict__ cap, const float* __restrict__ img,
                      _Float16* __restrict__ capn, _Float16* __restrict__ imgn) {
    __shared__ float sred[256];
    const int row = blockIdx.x;
    const int tid = threadIdx.x;
    const float* src = (blockIdx.y == 0) ? cap : img;
    _Float16* dst = (blockIdx.y == 0) ? capn : imgn;

    const float* r = src + (size_t)row * DDIM;
    float ss = 0.0f;
    #pragma unroll
    for (int c = tid; c < DDIM; c += 256) { float v = r[c]; ss += v * v; }
    sred[tid] = ss;
    __syncthreads();
    for (int s = 128; s > 0; s >>= 1) {
        if (tid < s) sred[tid] += sred[tid + s];
        __syncthreads();
    }
    const float scale = 1.0f / fmaxf(sqrtf(sred[0]), 1e-8f);
    _Float16* o = dst + (size_t)row * DDIM;
    #pragma unroll
    for (int c = tid; c < DDIM; c += 256) o[c] = (_Float16)(r[c] * scale);
}

// A-fragment fetch from LDS (two 16B ds loads -> one v16h)
__device__ __forceinline__ v16h afrag(const _Float16* aBase, int kk) {
    v8h lo = *(const v8h*)(aBase + kk * 32);
    v8h hi = *(const v8h*)(aBase + kk * 32 + 8);
    return __builtin_shufflevector(lo, hi, 0,1,2,3,4,5,6,7,8,9,10,11,12,13,14,15);
}

// ---------------------------------------------------------------------------
// Kernel 2: one InfoNCE direction, fixed-max LSE, 32 rows per wave.
// grid = (256 row-blocks of 32 rows, 2 col-halves), block = 128 (4 waves).
// Wave w covers cols [y*4096 + w*1024, +1024). Each B fragment feeds 2 WMMAs.
// Writes per-row partial (sumAll, sumPos) to ws (cols split across grid.y).
// ---------------------------------------------------------------------------
__global__ __launch_bounds__(128)
void gemm_lse_kernel(const _Float16* __restrict__ A, const _Float16* __restrict__ B,
                     const int* __restrict__ idx, float2* __restrict__ partials) {
    __shared__ _Float16 As[32 * LDS_STRIDE];
    __shared__ float2 red[4][32];

    const int rowBase = blockIdx.x * 32;
    const int tid  = threadIdx.x;
    const int wave = tid >> 5;
    const int lane = tid & 31;
    const int half16 = lane & 15;
    const int laneHi = lane >> 4;

    // ---- stage A tile (32 x 768 halves) into LDS, padded stride ----
    for (int i = tid; i < 32 * (DDIM / 8); i += 128) {
        int r = i / (DDIM / 8), c = i % (DDIM / 8);
        v8h v = *(const v8h*)(A + (size_t)(rowBase + r) * DDIM + c * 8);
        *(v8h*)&As[r * LDS_STRIDE + c * 8] = v;
    }
    __syncthreads();

    // row ids for this lane's 16 C-fragment rows (2 M-tiles x 8)
    int idxRow0[8], idxRow1[8];
    #pragma unroll
    for (int r = 0; r < 8; ++r) {
        idxRow0[r] = idx[rowBase +      laneHi * 8 + r];
        idxRow1[r] = idx[rowBase + 16 + laneHi * 8 + r];
    }

    float sAll0[8], sPos0[8], sAll1[8], sPos1[8];
    #pragma unroll
    for (int r = 0; r < 8; ++r) { sAll0[r]=0.f; sPos0[r]=0.f; sAll1[r]=0.f; sPos1[r]=0.f; }

    const int colBase = blockIdx.y * 4096 + wave * 1024;
    const _Float16* aBase0 = &As[(half16)      * LDS_STRIDE + laneHi * 16];
    const _Float16* aBase1 = &As[(16 + half16) * LDS_STRIDE + laneHi * 16];

    for (int ct = 0; ct < 64; ++ct) {
        const int col = colBase + ct * 16 + half16;
        const int idxCol = idx[col];
        const _Float16* bptr = B + (size_t)col * DDIM + laneHi * 16;

        v8f acc00 = {}, acc01 = {}, acc10 = {}, acc11 = {};
        #pragma unroll
        for (int kk = 0; kk < 24; kk += 2) {
            v16h b0 = *(const v16h*)(bptr + kk * 32);
            acc00 = __builtin_amdgcn_wmma_f32_16x16x32_f16(false, afrag(aBase0, kk), false, b0,
                                                           (short)0, acc00, false, false);
            acc10 = __builtin_amdgcn_wmma_f32_16x16x32_f16(false, afrag(aBase1, kk), false, b0,
                                                           (short)0, acc10, false, false);
            v16h b1 = *(const v16h*)(bptr + (kk + 1) * 32);
            acc01 = __builtin_amdgcn_wmma_f32_16x16x32_f16(false, afrag(aBase0, kk + 1), false, b1,
                                                           (short)0, acc01, false, false);
            acc11 = __builtin_amdgcn_wmma_f32_16x16x32_f16(false, afrag(aBase1, kk + 1), false, b1,
                                                           (short)0, acc11, false, false);
        }
        v8f acc0 = acc00 + acc01;
        v8f acc1 = acc10 + acc11;

        // fixed-shift exp fold: one transcendental per score element
        #pragma unroll
        for (int r = 0; r < 8; ++r) {
            float e0 = exp2f(fmaf(acc0[r], TEMP_SCALE, -MAXLOG));
            sAll0[r] += e0;
            sPos0[r] += (idxRow0[r] == idxCol) ? e0 : 0.0f;
            float e1 = exp2f(fmaf(acc1[r], TEMP_SCALE, -MAXLOG));
            sAll1[r] += e1;
            sPos1[r] += (idxRow1[r] == idxCol) ? e1 : 0.0f;
        }
    }

    // merge the 16 column residues (plain adds; fixed shared max)
    #pragma unroll
    for (int msk = 1; msk <= 8; msk <<= 1) {
        #pragma unroll
        for (int r = 0; r < 8; ++r) {
            sAll0[r] += __shfl_xor(sAll0[r], msk, 32);
            sPos0[r] += __shfl_xor(sPos0[r], msk, 32);
            sAll1[r] += __shfl_xor(sAll1[r], msk, 32);
            sPos1[r] += __shfl_xor(sPos1[r], msk, 32);
        }
    }
    if (half16 == 0) {   // lanes 0 / 16 hold merged rows
        #pragma unroll
        for (int r = 0; r < 8; ++r) {
            red[wave][laneHi * 8 + r]      = make_float2(sAll0[r], sPos0[r]);
            red[wave][16 + laneHi * 8 + r] = make_float2(sAll1[r], sPos1[r]);
        }
    }
    __syncthreads();

    // merge the 4 wave col-strips; write per-row partials for this col-half
    if (tid < 32) {
        float sA = 0.f, sP = 0.f;
        #pragma unroll
        for (int w = 0; w < 4; ++w) {
            float2 p = red[w][tid];
            sA += p.x;
            sP += p.y;
        }
        partials[(size_t)blockIdx.y * N_ROWS + rowBase + tid] = make_float2(sA, sP);
    }
}

// ---------------------------------------------------------------------------
// Kernel 3: per-row merge of the 2 col-halves, contribution, block tree-sum.
// grid = 64 blocks x 256 threads over 2 dirs * 8192 rows.
// ---------------------------------------------------------------------------
__global__ __launch_bounds__(256)
void row_reduce_kernel(const float2* __restrict__ partials, float* __restrict__ bp) {
    __shared__ float sred[256];
    const int tid = threadIdx.x;
    const int g = blockIdx.x * 256 + tid;       // 0 .. 16383
    const int dir = g >> 13;
    const int row = g & (N_ROWS - 1);
    const float2* p = partials + (size_t)dir * 2 * N_ROWS;
    float2 a = p[row];
    float2 b = p[N_ROWS + row];
    float c = LN2F * (log2f(a.x + b.x) - log2f(a.y + b.y));
    sred[tid] = c;
    __syncthreads();
    for (int k = 128; k > 0; k >>= 1) {
        if (tid < k) sred[tid] += sred[tid + k];
        __syncthreads();
    }
    if (tid == 0) bp[blockIdx.x] = sred[0];
}

// ---------------------------------------------------------------------------
// Kernel 4: deterministic final sum of 64 block partials.
// ---------------------------------------------------------------------------
__global__ __launch_bounds__(64)
void final_sum_kernel(const float* __restrict__ bp, float* __restrict__ out) {
    __shared__ float sred[64];
    const int tid = threadIdx.x;
    sred[tid] = bp[tid];
    __syncthreads();
    for (int k = 32; k > 0; k >>= 1) {
        if (tid < k) sred[tid] += sred[tid + k];
        __syncthreads();
    }
    if (tid == 0) out[0] = sred[0];
}

// ---------------------------------------------------------------------------
extern "C" void kernel_launch(void* const* d_in, const int* in_sizes, int n_in,
                              void* d_out, int out_size, void* d_ws, size_t ws_size,
                              hipStream_t stream) {
    (void)in_sizes; (void)n_in; (void)out_size; (void)ws_size;
    const float* cap = (const float*)d_in[0];
    const float* img = (const float*)d_in[1];
    const int*   idx = (const int*)d_in[2];
    float* out = (float*)d_out;

    char* ws = (char*)d_ws;
    const size_t matBytes = (size_t)N_ROWS * DDIM * sizeof(_Float16);
    _Float16* capn = (_Float16*)ws;
    _Float16* imgn = (_Float16*)(ws + matBytes);
    float2* partials = (float2*)(ws + 2 * matBytes);        // [2 dirs][2 halves][8192]
    float* bp = (float*)(ws + 2 * matBytes + (size_t)2 * 2 * N_ROWS * sizeof(float2));

    normalize_kernel<<<dim3(N_ROWS, 2), 256, 0, stream>>>(cap, img, capn, imgn);
    // direction 0: rows = captions, cols = images
    gemm_lse_kernel<<<dim3(256, 2), 128, 0, stream>>>(capn, imgn, idx, partials);
    // direction 1: transposed term = rows = images, cols = captions
    gemm_lse_kernel<<<dim3(256, 2), 128, 0, stream>>>(imgn, capn, idx, partials + 2 * N_ROWS);
    row_reduce_kernel<<<64, 256, 0, stream>>>(partials, bp);
    final_sum_kernel<<<1, 64, 0, stream>>>(bp, out);
}
